// MemoryModule_55894704390271
// MI455X (gfx1250) — matrix operand order repaired and checked by reading, not compiled, not run on hardware
//
#include <hip/hip_runtime.h>
#include <hip/hip_bf16.h>

// ---------------------------------------------------------------------------
// MemoryModule fused kernel for MI455X (gfx1250, wave32, WMMA)
//   q = l2norm(query); m = l2norm(memory)
//   att = softmax(q m^T); att = hardshrink-ish; att = l1norm(att)
//   z   = att @ memory
// Outputs concatenated: [z (32768x128) | att (32768x2000)] as f32.
// ---------------------------------------------------------------------------

#define N_Q   32768
#define DIM   128
#define M_MEM 2000
#define M_PAD 2016      // pad K of GEMM2 to multiple of 32
#define BM    32        // query rows per workgroup
#define SHRINK 0.0005f
#define EPSF   1e-12f

typedef __attribute__((ext_vector_type(16))) __bf16        v16bf;
typedef __attribute__((ext_vector_type(8)))  float         v8f;
typedef __attribute__((ext_vector_type(8)))  unsigned int  v8u;

__device__ __forceinline__ v8f wmma_bf16(v16bf a, v16bf b, v8f c) {
    // (neg_a, A, neg_b, B, c_mod, C, reuse_a, reuse_b)
    return __builtin_amdgcn_wmma_f32_16x16x32_bf16(false, a, false, b,
                                                   (short)0, c, false, false);
}

__device__ __forceinline__ float wave_sum(float v) {
    for (int m = 16; m >= 1; m >>= 1) v += __shfl_xor(v, m, 32);
    return v;
}
__device__ __forceinline__ float wave_max(float v) {
    for (int m = 16; m >= 1; m >>= 1) v = fmaxf(v, __shfl_xor(v, m, 32));
    return v;
}

// ---------------------------------------------------------------------------
// Prep: per memory row k (blocks 0..2015):
//   k < 2000: write l2-normalized row as bf16 hi/lo (GEMM1 B operand, row-major)
//             and raw row into transposed padded mT hi/lo (GEMM2 B operand)
//   k >= 2000: zero the padding columns of mT
// ---------------------------------------------------------------------------
__global__ __launch_bounds__(128)
void mm_prep_kernel(const float* __restrict__ memory,
                    __bf16* __restrict__ m_hi,  __bf16* __restrict__ m_lo,
                    __bf16* __restrict__ mT_hi, __bf16* __restrict__ mT_lo)
{
    const int k = blockIdx.x;    // memory row (K index of GEMM2)
    const int d = threadIdx.x;   // 0..127
    if (k >= M_MEM) {
        mT_hi[(size_t)d * M_PAD + k] = (__bf16)0.0f;
        mT_lo[(size_t)d * M_PAD + k] = (__bf16)0.0f;
        return;
    }
    const float x = memory[(size_t)k * DIM + d];

    // row sum of squares across 128 threads (4 waves)
    __shared__ float red[4];
    float ss = wave_sum(x * x);
    const int wv = d >> 5, ln = d & 31;
    if (ln == 0) red[wv] = ss;
    __syncthreads();
    const float tot = red[0] + red[1] + red[2] + red[3];
    const float inv = 1.0f / fmaxf(sqrtf(tot), EPSF);

    const float nx = x * inv;
    const __bf16 nh = (__bf16)nx;
    m_hi[(size_t)k * DIM + d] = nh;
    m_lo[(size_t)k * DIM + d] = (__bf16)(nx - (float)nh);

    const __bf16 rh = (__bf16)x;
    mT_hi[(size_t)d * M_PAD + k] = rh;
    mT_lo[(size_t)d * M_PAD + k] = (__bf16)(x - (float)rh);
}

// ---------------------------------------------------------------------------
// Fused main kernel: 1024 blocks x 256 threads (8 wave32s), 32 query rows each.
// ---------------------------------------------------------------------------
__global__ __launch_bounds__(256)
void mm_fused_kernel(const float* __restrict__ query,
                     const __bf16* __restrict__ m_hi,  const __bf16* __restrict__ m_lo,
                     const __bf16* __restrict__ mT_hi, const __bf16* __restrict__ mT_lo,
                     float* __restrict__ z_out, float* __restrict__ att_out)
{
    // s_att holds f32 attention scores through Phase C, then is overwritten in
    // place with packed {hi:bf16[31:16], lo:bf16[15:0]} dwords for Phase D.
    __shared__ __align__(32) unsigned int s_att[BM][M_PAD];   // 258048 B
    __shared__ __align__(32) __bf16       s_qhi[BM][DIM];     //   8192 B
    __shared__ __align__(32) __bf16       s_qlo[BM][DIM];     //   8192 B

    const int tid   = threadIdx.x;
    const int wave  = tid >> 5;
    const int lane  = tid & 31;
    const int row0  = blockIdx.x * BM;

    // ---- zero att padding columns [2000,2016): packed bf16 {0,0} == 0u ----
    for (int idx = tid; idx < BM * 16; idx += 256)
        s_att[idx >> 4][M_MEM + (idx & 15)] = 0u;

    // ---- Phase A: load & l2-normalize 32 query rows -> bf16 hi/lo ----
    for (int j = 0; j < 4; ++j) {
        const int r = wave * 4 + j;
        const float* qr = query + (size_t)(row0 + r) * DIM;
        float v[4];
        float ss = 0.0f;
        #pragma unroll
        for (int e = 0; e < 4; ++e) { v[e] = qr[lane * 4 + e]; ss += v[e] * v[e]; }
        ss = wave_sum(ss);
        const float inv = 1.0f / fmaxf(sqrtf(ss), EPSF);
        #pragma unroll
        for (int e = 0; e < 4; ++e) {
            const float x = v[e] * inv;
            const __bf16 h = (__bf16)x;
            s_qhi[r][lane * 4 + e] = h;
            s_qlo[r][lane * 4 + e] = (__bf16)(x - (float)h);
        }
    }
    __syncthreads();

    // Fragment geometry used here (C per ISA 7.12.2; A/B K order contiguous, not the ISA table order):
    //   A (16x32 bf16): lane L -> row M = L&15; lanes 0-15 hold K 0..15,
    //                   lanes 16-31 hold K 16..31
    //   B (32x16 bf16): lane L -> col N = L&15; 16 contiguous K, lanes>=16 offset +16
    //   C (16x16 f32):  VGPR r -> row M = r + 8*(L>>4); col N = L&15
    const int rt    = wave & 1;             // row tile (0/1) of the 32-row block
    const int arow  = rt * 16 + (lane & 15);
    const int khalf = lane >> 4;

    // ---- Phase B: GEMM1  s_att[32][2000] = qn @ mn^T  (bf16x3) ----
    for (int ct = (wave >> 1); ct < M_MEM / 16; ct += 4) {
        v8f acc = {};
        const int mrow = ct * 16 + (lane & 15);     // memory row = att column
        #pragma unroll
        for (int kc = 0; kc < DIM / 32; ++kc) {
            const int kbase = kc * 32;
            v16bf ahi, alo;
            #pragma unroll
            for (int e = 0; e < 8; ++e) {
                ahi[e]     = s_qhi[arow][kbase + khalf * 16 + e];
                ahi[e + 8] = s_qhi[arow][kbase + 8 + khalf * 16 + e];
                alo[e]     = s_qlo[arow][kbase + khalf * 16 + e];
                alo[e + 8] = s_qlo[arow][kbase + 8 + khalf * 16 + e];
            }
            const int kb = kbase + khalf * 16;
            const v16bf bhi = *(const v16bf*)(m_hi + (size_t)mrow * DIM + kb);
            const v16bf blo = *(const v16bf*)(m_lo + (size_t)mrow * DIM + kb);
            acc = wmma_bf16(ahi, bhi, acc);
            acc = wmma_bf16(ahi, blo, acc);
            acc = wmma_bf16(alo, bhi, acc);
        }
        #pragma unroll
        for (int r8 = 0; r8 < 8; ++r8)
            s_att[rt * 16 + r8 + 8 * khalf][ct * 16 + (lane & 15)] =
                __float_as_uint(acc[r8]);
    }
    __syncthreads();

    // ---- Phase C: softmax -> shrink -> l1-normalize ----
    // Streams final f32 att to HBM, and leaves packed bf16 hi/lo in LDS.
    for (int j = 0; j < 4; ++j) {
        const int r = wave * 4 + j;
        unsigned int* row = s_att[r];

        float mx = -3.402823466e38f;
        for (int k = lane; k < M_MEM; k += 32)
            mx = fmaxf(mx, __uint_as_float(row[k]));
        mx = wave_max(mx);

        float sum = 0.0f;
        for (int k = lane; k < M_MEM; k += 32) {
            const float e = __expf(__uint_as_float(row[k]) - mx);
            row[k] = __float_as_uint(e); sum += e;
        }
        sum = wave_sum(sum);
        const float invs = 1.0f / sum;

        float l1 = 0.0f;
        for (int k = lane; k < M_MEM; k += 32) {
            const float p = __uint_as_float(row[k]) * invs;
            const float d = p - SHRINK;
            const float t = fmaxf(d, 0.0f) * p / (fabsf(d) + EPSF);
            row[k] = __float_as_uint(t); l1 += t;          // t >= 0
        }
        l1 = wave_sum(l1);
        const float sc = 1.0f / fmaxf(l1, EPSF);

        float* gout = att_out + (size_t)(row0 + r) * M_MEM;
        for (int k = lane; k < M_MEM; k += 32) {
            const float v = __uint_as_float(row[k]) * sc;
            gout[k] = v;                                    // f32 att to HBM
            const __bf16 h = (__bf16)v;                     // pack bf16 hi/lo
            const __bf16 l = (__bf16)(v - (float)h);
            const unsigned int hb = __builtin_bit_cast(unsigned short, h);
            const unsigned int lb = __builtin_bit_cast(unsigned short, l);
            row[k] = (hb << 16) | lb;                       // {hi[31:16], lo[15:0]}
        }
    }
    __syncthreads();

    // ---- Phase D: GEMM2  z[32][128] = att @ memory  (bf16x3, K padded) ----
    // A fragments built from packed LDS dwords with one v_perm_b32 per dword:
    //   V_PERM_B32(S0,S1,sel): S1 supplies byte indices 0-3, S0 indices 4-7.
    //   hi pair {w0.hi, w1.hi} -> sel 0x07060302 ; lo pair -> sel 0x05040100
    const unsigned int* rowp = &s_att[arow][0];
    #pragma unroll
    for (int cti = 0; cti < 2; ++cti) {
        const int ct   = (wave >> 1) + cti * 4;      // D-tile (8 tiles of 16)
        const int ncol = ct * 16 + (lane & 15);      // z column = mT row
        v8f acc = {};
        for (int kc = 0; kc < M_PAD / 32; ++kc) {
            const int kbase = kc * 32;
            const v8u w0 = *(const v8u*)(rowp + kbase + khalf * 16);
            const v8u w1 = *(const v8u*)(rowp + kbase + 8 + khalf * 16);
            v8u hu, lu;
            #pragma unroll
            for (int e = 0; e < 4; ++e) {
                hu[e]     = __builtin_amdgcn_perm(w0[2*e + 1], w0[2*e], 0x07060302u);
                lu[e]     = __builtin_amdgcn_perm(w0[2*e + 1], w0[2*e], 0x05040100u);
                hu[e + 4] = __builtin_amdgcn_perm(w1[2*e + 1], w1[2*e], 0x07060302u);
                lu[e + 4] = __builtin_amdgcn_perm(w1[2*e + 1], w1[2*e], 0x05040100u);
            }
            const v16bf ahi = __builtin_bit_cast(v16bf, hu);
            const v16bf alo = __builtin_bit_cast(v16bf, lu);
            const int kb = kbase + khalf * 16;
            const v16bf bhi = *(const v16bf*)(mT_hi + (size_t)ncol * M_PAD + kb);
            const v16bf blo = *(const v16bf*)(mT_lo + (size_t)ncol * M_PAD + kb);
            acc = wmma_bf16(ahi, bhi, acc);
            acc = wmma_bf16(ahi, blo, acc);
            acc = wmma_bf16(alo, bhi, acc);
        }
        #pragma unroll
        for (int r8 = 0; r8 < 8; ++r8) {
            const int grow = row0 + rt * 16 + r8 + 8 * khalf;
            z_out[(size_t)grow * DIM + ct * 16 + (lane & 15)] = acc[r8];
        }
    }
}

// ---------------------------------------------------------------------------
extern "C" void kernel_launch(void* const* d_in, const int* in_sizes, int n_in,
                              void* d_out, int out_size, void* d_ws, size_t ws_size,
                              hipStream_t stream) {
    (void)in_sizes; (void)n_in; (void)out_size; (void)ws_size;
    const float* query  = (const float*)d_in[0];   // [32768,128]
    const float* memory = (const float*)d_in[1];   // [2000,128]

    // workspace layout (bytes)
    char* ws = (char*)d_ws;
    __bf16* m_hi  = (__bf16*)(ws);                                  // 2000*128*2 = 512000
    __bf16* m_lo  = (__bf16*)(ws + 512000);                         // 512000
    __bf16* mT_hi = (__bf16*)(ws + 1024000);                        // 128*2016*2 = 516096
    __bf16* mT_lo = (__bf16*)(ws + 1540096);                        // 516096  (total ~2.06 MB)

    float* z_out   = (float*)d_out;                 // [32768,128]
    float* att_out = z_out + (size_t)N_Q * DIM;     // [32768,2000]

    mm_prep_kernel<<<M_PAD, 128, 0, stream>>>(memory, m_hi, m_lo, mT_hi, mT_lo);
    mm_fused_kernel<<<N_Q / BM, 256, 0, stream>>>(query, m_hi, m_lo, mT_hi, mT_lo,
                                                  z_out, att_out);
}
